// LogGeodesicLoss_70746701300160
// MI455X (gfx1250) — matrix operand order repaired and checked by reading, not compiled, not run on hardware
//
#include <hip/hip_runtime.h>
#include <hip/hip_bf16.h>
#include <math.h>

typedef __attribute__((ext_vector_type(16))) _Float16 v16h;
typedef __attribute__((ext_vector_type(8)))  _Float16 v8h;
typedef __attribute__((ext_vector_type(8)))  float    v8f;

#define WAVES 8   // waves per 256-thread block, one matrix per wave

// ---------------------------------------------------------------------------
// WMMA fragment loaders from LDS (row-major 32x32 f16, symmetric matrices).
// A-frag (16x32, tile row mt): per CDNA5 ISA 16-bit A layout:
//   lane L<16 : M=mt*16+L,   halves = K[0..7] ++ K[16..23]
//   lane L>=16: M=mt*16+L-16,halves = K[8..15] ++ K[24..31]
// => two contiguous 16B chunks per lane.
// B-frag (32x16, tile col nt): lanes 0-15 hold N=lane, K[0..7] ++ K[16..23];
//   lanes 16-31 hold N=lane-16, K[8..15] ++ K[24..31]. Column access of B == row access
//   of B^T == row access of B (symmetric) => two contiguous 16B chunks per lane.
// ---------------------------------------------------------------------------
static __device__ __forceinline__ v16h load_a_frag(const _Float16* buf, int mt, int lane) {
    int M  = mt * 16 + (lane & 15);
    int k0 = (lane >> 4) * 8;
    union { v16h v; v8h h[2]; } u;
    u.h[0] = *(const v8h*)(buf + M * 32 + k0);
    u.h[1] = *(const v8h*)(buf + M * 32 + k0 + 16);
    return u.v;
}

static __device__ __forceinline__ v16h load_b_frag(const _Float16* buf, int nt, int lane) {
    int N  = nt * 16 + (lane & 15);
    int K0 = (lane >> 4) * 8;
    union { v16h v; v8h h[2]; } u;
    u.h[0] = *(const v8h*)(buf + N * 32 + K0);
    u.h[1] = *(const v8h*)(buf + N * 32 + K0 + 16);
    return u.v;
}

// C/D f32 16x16 layout: reg r: lanes 0-15 -> (M=r, N=lane); lanes 16-31 ->
// (M=r+8, N=lane-16). Store transposed (valid: result is symmetric) so that
// consecutive regs hit consecutive halves -> packed ds_store_b32.
// out = alpha * (A*B) + diag * I
static __device__ __forceinline__ void store_c_tile(_Float16* buf, int mt, int nt, int lane,
                                                    v8f c, float alpha, float diag) {
    int n     = nt * 16 + (lane & 15);
    int mbase = mt * 16 + 8 * (lane >> 4);
#pragma unroll
    for (int rp = 0; rp < 4; ++rp) {
        int m0 = mbase + 2 * rp;
        float v0 = alpha * c[2 * rp]     + ((m0     == n) ? diag : 0.0f);
        float v1 = alpha * c[2 * rp + 1] + ((m0 + 1 == n) ? diag : 0.0f);
        union { unsigned u; _Float16 h[2]; } p;
        p.h[0] = (_Float16)v0;
        p.h[1] = (_Float16)v1;
        *(unsigned*)(buf + n * 32 + m0) = p.u;
    }
}

// 32x32 GEMM on one wave: out = alpha*(A*B) + diag*I.  4x v_wmma_f32_16x16x32_f16.
// Safe for out aliasing A or B: all LDS loads are issued before all stores and
// same-wave LDS ops are in order.
static __device__ __forceinline__ void matmul32(_Float16* out, const _Float16* A,
                                                const _Float16* Bm, int lane,
                                                float alpha, float diag) {
    v16h a0 = load_a_frag(A, 0, lane);
    v16h a1 = load_a_frag(A, 1, lane);
    v16h b0 = load_b_frag(Bm, 0, lane);
    v16h b1 = load_b_frag(Bm, 1, lane);
    v8f z = {};
    v8f c00 = __builtin_amdgcn_wmma_f32_16x16x32_f16(false, a0, false, b0, (short)0, z, false, false);
    v8f c01 = __builtin_amdgcn_wmma_f32_16x16x32_f16(false, a0, false, b1, (short)0, z, false, false);
    v8f c10 = __builtin_amdgcn_wmma_f32_16x16x32_f16(false, a1, false, b0, (short)0, z, false, false);
    v8f c11 = __builtin_amdgcn_wmma_f32_16x16x32_f16(false, a1, false, b1, (short)0, z, false, false);
    store_c_tile(out, 0, 0, lane, c00, alpha, diag);
    store_c_tile(out, 0, 1, lane, c01, alpha, diag);
    store_c_tile(out, 1, 0, lane, c10, alpha, diag);
    store_c_tile(out, 1, 1, lane, c11, alpha, diag);
}

// logm of one 32x32 SPD matrix (eigenvalues >= 1) on one wave.
// Inverse scaling & squaring: 5 Newton-Schulz square roots + 4-term log series.
// Result returned per-lane: F[j] = logm(A)[j][lane].
static __device__ void logm32(const float* gA, _Float16* Y, _Float16* Z, _Float16* T,
                              int lane, float* F) {
    // c = inf-norm (upper bound of lambda_max for symmetric A)
    float s = 0.0f;
    const float* row = gA + lane * 32;
#pragma unroll
    for (int j = 0; j < 32; ++j) s += fabsf(row[j]);
#pragma unroll
    for (int m = 16; m >= 1; m >>= 1) s = fmaxf(s, __shfl_xor(s, m, 32));
    float c = s, invc = 1.0f / s;

    // Y = A/c (eigenvalues in (0,1]), Z = I
    for (int i = lane; i < 1024; i += 32) {
        Y[i] = (_Float16)(gA[i] * invc);
        Z[i] = (_Float16)(((i >> 5) == (i & 31)) ? 1.0f : 0.0f);
    }

    // 5 square roots: Y <- Y^(1/2) via coupled Newton-Schulz
    const int iters[5] = {8, 7, 6, 5, 4};
#pragma unroll
    for (int rt = 0; rt < 5; ++rt) {
        if (rt) {
            for (int i = lane; i < 1024; i += 32)
                Z[i] = (_Float16)(((i >> 5) == (i & 31)) ? 1.0f : 0.0f);
        }
        for (int k = 0; k < iters[rt]; ++k) {
            matmul32(T, Z, Y, lane, -0.5f, 1.5f);  // T = (3I - Z*Y)/2
            matmul32(Y, Y, T, lane, 1.0f, 0.0f);   // Y = Y*T
            matmul32(Z, T, Z, lane, 1.0f, 0.0f);   // Z = T*Z
        }
    }

    // M = Y - I (in Z), S3 = I/3 - M/4 (in T);  ||M|| <= ~0.06
    for (int i = lane; i < 1024; i += 32) {
        float d = ((i >> 5) == (i & 31)) ? 1.0f : 0.0f;
        float m = (float)Y[i] - d;
        Z[i] = (_Float16)m;
        T[i] = (_Float16)((1.0f / 3.0f) * d - 0.25f * m);
    }
    matmul32(Y, Z, T, lane, 1.0f, -0.5f);  // S2 = M*S3 - I/2
    matmul32(T, Z, Y, lane, 1.0f, 1.0f);   // S1 = M*S2 + I
    matmul32(Y, Z, T, lane, 1.0f, 0.0f);   // L  = M*S1 ~ log(A^(1/32) / c^(1/32))... (log of scaled root)

    // logm(A) = 2^5 * L + log(c) * I
    float lc = logf(c);
#pragma unroll
    for (int j = 0; j < 32; ++j)
        F[j] = 32.0f * (float)Y[j * 32 + lane] + ((j == lane) ? lc : 0.0f);
}

__global__ __launch_bounds__(256) void LogGeodesicLoss_main_kernel(
        const float* __restrict__ yhat, const float* __restrict__ yy,
        float* __restrict__ ws, int batch) {
    __shared__ __align__(16) _Float16 smem[WAVES][3][1024];
    const int lane = threadIdx.x & 31;
    const int w    = threadIdx.x >> 5;
    const int b    = blockIdx.x * WAVES + w;
    if (b >= batch) return;  // wave-uniform; EXEC stays all-ones for WMMA

    _Float16* Y = smem[w][0];
    _Float16* Z = smem[w][1];
    _Float16* T = smem[w][2];

    float F1[32], F2[32];
    logm32(yhat + (size_t)b * 1024, Y, Z, T, lane, F1);
    logm32(yy   + (size_t)b * 1024, Y, Z, T, lane, F2);

    // tr(D*D) = ||D||_F^2 (D symmetric)
    float acc = 0.0f;
#pragma unroll
    for (int j = 0; j < 32; ++j) {
        float d = F1[j] - F2[j];
        acc += d * d;
    }
#pragma unroll
    for (int m = 16; m >= 1; m >>= 1) acc += __shfl_xor(acc, m, 32);
    if (lane == 0) ws[b] = sqrtf(acc);
}

// Deterministic final reduction (fixed tree, no atomics -> replay-stable).
__global__ __launch_bounds__(256) void LogGeodesicLoss_reduce_kernel(
        const float* __restrict__ ws, float* __restrict__ out, int n) {
    __shared__ float sm[256];
    float s = 0.0f;
    for (int i = threadIdx.x; i < n; i += 256) s += ws[i];
    sm[threadIdx.x] = s;
    __syncthreads();
    for (int k = 128; k > 0; k >>= 1) {
        if ((int)threadIdx.x < k) sm[threadIdx.x] += sm[threadIdx.x + k];
        __syncthreads();
    }
    if (threadIdx.x == 0) out[0] = sm[0];
}

extern "C" void kernel_launch(void* const* d_in, const int* in_sizes, int n_in,
                              void* d_out, int out_size, void* d_ws, size_t ws_size,
                              hipStream_t stream) {
    const float* yhat = (const float*)d_in[0];
    const float* yy   = (const float*)d_in[1];
    const int batch   = in_sizes[0] / 1024;  // 16384 matrices of 32x32

    float* ws = (float*)d_ws;  // batch floats of per-sample sqrt(tr)

    dim3 grid((batch + WAVES - 1) / WAVES);
    LogGeodesicLoss_main_kernel<<<grid, 256, 0, stream>>>(yhat, yy, ws, batch);
    LogGeodesicLoss_reduce_kernel<<<1, 256, 0, stream>>>(ws, (float*)d_out, batch);
}
